// MultiHeadSelfAttention_74448963109499
// MI455X (gfx1250) — matrix operand order repaired and checked by reading, not compiled, not run on hardware
//
#include <hip/hip_runtime.h>
#include <hip/hip_bf16.h>
#include <math.h>

// ---------------------------------------------------------------------------
// CDNA5 (gfx1250) wave32 WMMA types
// ---------------------------------------------------------------------------
typedef __attribute__((ext_vector_type(16))) __bf16    v16bf;
typedef __attribute__((ext_vector_type(2)))  __bf16    v2bf;
typedef __attribute__((ext_vector_type(8)))  float     v8f;
typedef __attribute__((ext_vector_type(8)))  unsigned  v8u;

// fp32 -> bf16 (round-half-up; 2 VALU ops)
__device__ __forceinline__ unsigned short f32_to_bf16(float f) {
    return (unsigned short)((__float_as_uint(f) + 0x8000u) >> 16);
}

// Pack two fp32 -> packed bf16x2 in one u32.
// Preferred: native v_cvt_pk_bf16_f32 (1 op, RNE) if the toolchain has it.
// Fallback: two rounding adds + one v_perm_b32 byte-select (3 ops total).
__device__ __forceinline__ unsigned pack_bf16x2(float lo, float hi) {
#if __has_builtin(__builtin_amdgcn_cvt_pk_bf16_f32)
    v2bf p = __builtin_amdgcn_cvt_pk_bf16_f32(lo, hi);
    return __builtin_bit_cast(unsigned, p);
#else
    unsigned a = __float_as_uint(lo) + 0x8000u;
    unsigned b = __float_as_uint(hi) + 0x8000u;
    // result bytes [3:0] = {b[3], b[2], a[3], a[2]}  ->  {hi_bf16, lo_bf16}
    return __builtin_amdgcn_perm(b, a, 0x07060302u);
#endif
}

// A-operand K-pair base for VGPR v (ISA 05_wmma 16-bit A 16x32 layout):
// VGPRs 0-3 hold K pairs {0,1}..{6,7}, VGPRs 4-7 hold {16,17}..{22,23};
// lane-half g (lane>=16) adds +8.
__device__ __forceinline__ int a_kpair(int v, int g) {
    return (v < 4 ? 2 * v : 16 + 2 * (v - 4)) + 8 * g;
}
// B-operand K-pair base for VGPR v: same as A (VGPRs 0-3 {0,1}..{6,7}, 4-7 {16,17}..{22,23}), lane-half adds +8.
__device__ __forceinline__ int b_kpair(int v, int g) {
    return (v < 4 ? 2 * v : 16 + 2 * (v - 4)) + 8 * g;
}

__device__ __forceinline__ v16bf bits_to_bf16(v8u u) {
    return __builtin_bit_cast(v16bf, u);
}

__device__ __forceinline__ float rowmax16(float v) {
    #pragma unroll
    for (int m = 8; m >= 1; m >>= 1) v = fmaxf(v, __shfl_xor(v, m, 16));
    return v;
}
__device__ __forceinline__ float rowsum16(float v) {
    #pragma unroll
    for (int m = 8; m >= 1; m >>= 1) v += __shfl_xor(v, m, 16);
    return v;
}

// ---------------------------------------------------------------------------
// Kernel 1: C[M,N] = A[M,K] @ W[K,N] + bias   (fp32 in/out, bf16 WMMA compute)
// Block: 256 threads (8 waves). Tile: 128(M) x 128(N), K-step 32.
// Software-pipelined: next K-tile is loaded to registers while the current
// tile (double-buffered in LDS) feeds 64 WMMAs; one barrier per K-step.
// ---------------------------------------------------------------------------
#define GT_M 128
#define GT_N 128
#define GT_K 32

__global__ __launch_bounds__(256) void gemm_bias_kernel(
    const float* __restrict__ A, const float* __restrict__ W,
    const float* __restrict__ bias, float* __restrict__ C,
    int M, int N, int K)
{
    __shared__ unsigned short As[2][GT_M * GT_K];   // [row][k]        2 x 8 KB
    __shared__ unsigned short Bs[2][GT_N * GT_K];   // [n][k] (transp) 2 x 8 KB

    const int tid  = threadIdx.x;
    const int wave = tid >> 5;
    const int lane = tid & 31;
    const int L    = lane & 15;
    const int g    = lane >> 4;
    const int tileM = blockIdx.y * GT_M;
    const int tileN = blockIdx.x * GT_N;

    v8f acc[8];
    #pragma unroll
    for (int t = 0; t < 8; ++t) acc[t] = (v8f)(0.0f);

    // ---- prologue: load K-tile 0 into registers (4 float4 of A + of W)
    float4 fa[4], fw[4];
    #pragma unroll
    for (int i = 0; i < 4; ++i) {
        int idx4 = tid + i * 256;
        fa[i] = *(const float4*)&A[(size_t)(tileM + (idx4 >> 3)) * K + (idx4 & 7) * 4];
        fw[i] = *(const float4*)&W[(size_t)(idx4 >> 5) * N + tileN + (idx4 & 31) * 4];
    }

    int buf = 0;
    for (int k0 = 0; k0 < K; k0 += GT_K, buf ^= 1) {
        // ---- pack current regs -> LDS[buf]
        #pragma unroll
        for (int i = 0; i < 4; ++i) {
            int idx4 = tid + i * 256;
            uint2 p = make_uint2(pack_bf16x2(fa[i].x, fa[i].y),
                                 pack_bf16x2(fa[i].z, fa[i].w));
            *(uint2*)&As[buf][(idx4 >> 3) * GT_K + (idx4 & 7) * 4] = p;
            unsigned p01 = pack_bf16x2(fw[i].x, fw[i].y);
            unsigned p23 = pack_bf16x2(fw[i].z, fw[i].w);
            int kk = idx4 >> 5, nn4 = (idx4 & 31) * 4;
            Bs[buf][(nn4 + 0) * GT_K + kk] = (unsigned short)p01;
            Bs[buf][(nn4 + 1) * GT_K + kk] = (unsigned short)(p01 >> 16);
            Bs[buf][(nn4 + 2) * GT_K + kk] = (unsigned short)p23;
            Bs[buf][(nn4 + 3) * GT_K + kk] = (unsigned short)(p23 >> 16);
        }
        // ---- issue next tile's global loads (latency hidden by compute)
        if (k0 + GT_K < K) {
            #pragma unroll
            for (int i = 0; i < 4; ++i) {
                int idx4 = tid + i * 256;
                fa[i] = *(const float4*)&A[(size_t)(tileM + (idx4 >> 3)) * K + k0 + GT_K + (idx4 & 7) * 4];
                fw[i] = *(const float4*)&W[(size_t)(k0 + GT_K + (idx4 >> 5)) * N + tileN + (idx4 & 31) * 4];
            }
        }
        __syncthreads();

        // ---- A fragment for this wave (row = wave*16 + L)
        v8u au;
        const int arow = (wave * 16 + L) * GT_K;
        #pragma unroll
        for (int v = 0; v < 8; ++v)
            au[v] = *(const unsigned*)&As[buf][arow + a_kpair(v, g)];
        v16bf af = bits_to_bf16(au);

        // ---- 8 B fragments + WMMA accumulate
        #pragma unroll
        for (int t = 0; t < 8; ++t) {
            v8u bu;
            const int nrow = (t * 16 + L) * GT_K;
            #pragma unroll
            for (int v = 0; v < 8; ++v)
                bu[v] = *(const unsigned*)&Bs[buf][nrow + b_kpair(v, g)];
            v16bf bf = bits_to_bf16(bu);
            acc[t] = __builtin_amdgcn_wmma_f32_16x16x32_bf16(
                false, af, false, bf, (short)0, acc[t], false, false);
        }
        // no trailing barrier: next iteration writes the other LDS buffer,
        // and its barrier orders those writes against all readers.
    }

    // ---- epilogue: C-layout lane holds col n = L, rows m = r + 8*g
    #pragma unroll
    for (int t = 0; t < 8; ++t) {
        int n = tileN + t * 16 + L;
        float bv = bias[n];
        #pragma unroll
        for (int r = 0; r < 8; ++r) {
            int m = tileM + wave * 16 + r + 8 * g;
            C[(size_t)m * N + n] = acc[t][r] + bv;
        }
    }
}

// ---------------------------------------------------------------------------
// Kernel 2: causal flash attention over qkv workspace.
// qkv: [B*T, 3072] with cols [0,1024)=Q, [1024,2048)=K, [2048,3072)=V.
// out: [B*T, 1024] (pre-projection attention output).
// Grid: (B*H, T/128). Block: 256 threads = 8 waves; wave w owns 16 q rows.
// K/V tiles are register-pipelined + double-buffered; one barrier per step.
// ---------------------------------------------------------------------------
#define AT_T   2048
#define AT_C3  3072
#define AT_C   1024
#define AT_DH  64

__global__ __launch_bounds__(256) void attn_kernel(
    const float* __restrict__ qkv, float* __restrict__ out)
{
    const int bh   = blockIdx.x;          // b*16 + h
    const int b    = bh >> 4;
    const int h    = bh & 15;
    const int qblk = blockIdx.y;          // 128 q-rows per block
    const int tid  = threadIdx.x;
    const int wave = tid >> 5;
    const int lane = tid & 31;
    const int L    = lane & 15;
    const int g    = lane >> 4;
    const int qbase = qblk * 128 + wave * 16;          // wave's first q row

    const float* base = qkv + (size_t)b * AT_T * AT_C3;

    __shared__ unsigned short Ks[2][32 * 64];   // [key][dh]  bf16, 2 x 4 KB
    __shared__ unsigned short Vt[2][64 * 32];   // [dh][key]  bf16, 2 x 4 KB
    __shared__ unsigned short Ps[8][16 * 32];   // per-wave P tile, 8 KB

    // ---- preload Q as two bf16 A-fragments (dh chunks of 32)
    v16bf qa[2];
    {
        const float* qr = base + (size_t)(qbase + L) * AT_C3 + h * AT_DH;
        #pragma unroll
        for (int c = 0; c < 2; ++c) {
            v8u au;
            #pragma unroll
            for (int h2 = 0; h2 < 2; ++h2) {           // VGPRs 0-3 / 4-7
                int kb8 = 32 * c + 16 * h2 + 8 * g;    // 8 consecutive k
                float4 f0 = *(const float4*)&qr[kb8];
                float4 f1 = *(const float4*)&qr[kb8 + 4];
                au[4 * h2 + 0] = pack_bf16x2(f0.x, f0.y);
                au[4 * h2 + 1] = pack_bf16x2(f0.z, f0.w);
                au[4 * h2 + 2] = pack_bf16x2(f1.x, f1.y);
                au[4 * h2 + 3] = pack_bf16x2(f1.z, f1.w);
            }
            qa[c] = bits_to_bf16(au);
        }
    }

    v8f O[4];
    #pragma unroll
    for (int t = 0; t < 4; ++t) O[t] = (v8f)(0.0f);
    float mrun[8], lrun[8];
    #pragma unroll
    for (int r = 0; r < 8; ++r) { mrun[r] = -INFINITY; lrun[r] = 0.0f; }

    const int steps = (qblk + 1) * 4;            // 32-key steps (causal bound)

    // ---- prologue: load K/V tile 0 into registers (2 float4 each)
    float4 fk[2], fv[2];
    #pragma unroll
    for (int i = 0; i < 2; ++i) {
        int idx4 = tid + i * 256;
        const float* kv = base + (size_t)(idx4 >> 4) * AT_C3 + h * AT_DH + (idx4 & 15) * 4;
        fk[i] = *(const float4*)&kv[AT_C];
        fv[i] = *(const float4*)&kv[2 * AT_C];
    }

    int buf = 0;
    for (int s = 0; s < steps; ++s, buf ^= 1) {
        const int kb = s * 32;
        // ---- pack current K/V regs -> LDS[buf]
        #pragma unroll
        for (int i = 0; i < 2; ++i) {
            int idx4 = tid + i * 256;
            int key = idx4 >> 4, dh4 = (idx4 & 15) * 4;
            uint2 pk = make_uint2(pack_bf16x2(fk[i].x, fk[i].y),
                                  pack_bf16x2(fk[i].z, fk[i].w));
            *(uint2*)&Ks[buf][key * 64 + dh4] = pk;
            unsigned pv01 = pack_bf16x2(fv[i].x, fv[i].y);
            unsigned pv23 = pack_bf16x2(fv[i].z, fv[i].w);
            Vt[buf][(dh4 + 0) * 32 + key] = (unsigned short)pv01;
            Vt[buf][(dh4 + 1) * 32 + key] = (unsigned short)(pv01 >> 16);
            Vt[buf][(dh4 + 2) * 32 + key] = (unsigned short)pv23;
            Vt[buf][(dh4 + 3) * 32 + key] = (unsigned short)(pv23 >> 16);
        }
        // ---- issue next tile's loads (hidden behind softmax + WMMA)
        if (s + 1 < steps) {
            #pragma unroll
            for (int i = 0; i < 2; ++i) {
                int idx4 = tid + i * 256;
                const float* kv = base + (size_t)(kb + 32 + (idx4 >> 4)) * AT_C3 + h * AT_DH + (idx4 & 15) * 4;
                fk[i] = *(const float4*)&kv[AT_C];
                fv[i] = *(const float4*)&kv[2 * AT_C];
            }
        }
        __syncthreads();

        // ---- S = Q @ K^T : two 16x16 tiles, dh reduced in two 32-chunks
        v8f S[2];
        #pragma unroll
        for (int j = 0; j < 2; ++j) {
            S[j] = (v8f)(0.0f);
            #pragma unroll
            for (int c = 0; c < 2; ++c) {
                v8u bu;
                const int krow = (j * 16 + L) * 64 + 32 * c;
                #pragma unroll
                for (int v = 0; v < 8; ++v)
                    bu[v] = *(const unsigned*)&Ks[buf][krow + b_kpair(v, g)];
                S[j] = __builtin_amdgcn_wmma_f32_16x16x32_bf16(
                    false, qa[c], false, bits_to_bf16(bu), (short)0, S[j], false, false);
            }
        }

        // ---- scale + causal mask (analytic; lane col = key, row = r + 8g)
        const float sc = 0.125f;   // 1/sqrt(64)
        #pragma unroll
        for (int j = 0; j < 2; ++j) {
            const int key = kb + j * 16 + L;
            #pragma unroll
            for (int r = 0; r < 8; ++r) {
                const int qg = qbase + r + 8 * g;
                S[j][r] = (key <= qg) ? S[j][r] * sc : -INFINITY;
            }
        }

        // ---- online softmax: row stats live in accumulator-aligned form
        float mnew[8], fac[8];
        #pragma unroll
        for (int r = 0; r < 8; ++r) {
            float rm = rowmax16(fmaxf(S[0][r], S[1][r]));
            mnew[r] = fmaxf(mrun[r], rm);
            fac[r]  = __expf(mrun[r] - mnew[r]);
            mrun[r] = mnew[r];
        }
        float P0[8], P1[8];
        #pragma unroll
        for (int r = 0; r < 8; ++r) {
            P0[r] = __expf(S[0][r] - mnew[r]);
            P1[r] = __expf(S[1][r] - mnew[r]);
            float rs = rowsum16(P0[r] + P1[r]);
            lrun[r] = lrun[r] * fac[r] + rs;
            #pragma unroll
            for (int t = 0; t < 4; ++t) O[t][r] *= fac[r];
        }

        // ---- re-layout P (C-form -> A-form) through per-wave LDS
        #pragma unroll
        for (int r = 0; r < 8; ++r) {
            const int row = r + 8 * g;
            unsigned p = pack_bf16x2(P0[r], P1[r]);
            Ps[wave][row * 32 + L]      = (unsigned short)p;
            Ps[wave][row * 32 + 16 + L] = (unsigned short)(p >> 16);
        }
        asm volatile("s_wait_dscnt 0" ::: "memory");   // same-wave LDS RAW

        v8u au;
        #pragma unroll
        for (int v = 0; v < 8; ++v)
            au[v] = *(const unsigned*)&Ps[wave][L * 32 + a_kpair(v, g)];
        v16bf pa = bits_to_bf16(au);

        // ---- O += P @ V (V transposed in LDS: key pairs contiguous)
        #pragma unroll
        for (int t = 0; t < 4; ++t) {
            v8u bu;
            const int vrow = (t * 16 + L) * 32;
            #pragma unroll
            for (int v = 0; v < 8; ++v)
                bu[v] = *(const unsigned*)&Vt[buf][vrow + b_kpair(v, g)];
            O[t] = __builtin_amdgcn_wmma_f32_16x16x32_bf16(
                false, pa, false, bits_to_bf16(bu), (short)0, O[t], false, false);
        }
        // no trailing barrier: next iteration stages the other K/V buffer.
    }

    // ---- epilogue: out[b*T+q][h*64 + dh] = O * (1/l)
    float inv[8];
    #pragma unroll
    for (int r = 0; r < 8; ++r) inv[r] = 1.0f / lrun[r];
    #pragma unroll
    for (int t = 0; t < 4; ++t) {
        const int dh = t * 16 + L;
        #pragma unroll
        for (int r = 0; r < 8; ++r) {
            const int qg = qbase + r + 8 * g;
            out[((size_t)b * AT_T + qg) * AT_C + h * AT_DH + dh] = O[t][r] * inv[r];
        }
    }
}

// ---------------------------------------------------------------------------
// Host-side launcher
// ---------------------------------------------------------------------------
extern "C" void kernel_launch(void* const* d_in, const int* in_sizes, int n_in,
                              void* d_out, int out_size, void* d_ws, size_t ws_size,
                              hipStream_t stream) {
    (void)in_sizes; (void)n_in; (void)out_size; (void)ws_size;
    const float* x     = (const float*)d_in[0];   // [2,2048,1024]
    const float* Wqkv  = (const float*)d_in[1];   // [1024,3072]
    const float* bqkv  = (const float*)d_in[2];   // [3072]
    const float* Wproj = (const float*)d_in[3];   // [1024,1024]
    const float* bproj = (const float*)d_in[4];   // [1024]
    // d_in[5] causal mask: applied analytically in attn_kernel.

    const int M = 2 * 2048;   // B*T
    float* qkv = (float*)d_ws;                       // [4096,3072]
    float* att = qkv + (size_t)M * 3072;             // [4096,1024]
    float* out = (float*)d_out;                      // [4096,1024]

    dim3 blk(256);
    // QKV projection: [4096,1024] @ [1024,3072] + b
    gemm_bias_kernel<<<dim3(3072 / GT_N, M / GT_M), blk, 0, stream>>>(
        x, Wqkv, bqkv, qkv, M, 3072, 1024);
    // Causal flash attention per (b,h), 128 q-rows per block
    attn_kernel<<<dim3(32, AT_T / 128), blk, 0, stream>>>(qkv, att);
    // Output projection: [4096,1024] @ [1024,1024] + b
    gemm_bias_kernel<<<dim3(1024 / GT_N, M / GT_M), blk, 0, stream>>>(
        att, Wproj, bproj, out, M, 1024, 1024);
}
